// GAT_14362370638270
// MI455X (gfx1250) — matrix operand order repaired and checked by reading, not compiled, not run on hardware
//
#include <hip/hip_runtime.h>
#include <math.h>

typedef __attribute__((ext_vector_type(16))) _Float16 v16h;
typedef __attribute__((ext_vector_type(8)))  float    v8f;

#define NEG_SLOPE 0.2f

// ---------- float <-> ordered-uint encoding for atomicMax on floats ----------
__device__ __forceinline__ unsigned ford(float f) {
    unsigned u = __float_as_uint(f);
    return (u & 0x80000000u) ? ~u : (u | 0x80000000u);
}
__device__ __forceinline__ float funord(unsigned u) {
    return __uint_as_float((u & 0x80000000u) ? (u & 0x7FFFFFFFu) : ~u);
}
#define ORD_NEG_INF 0x007FFFFFu   // ford(-INF)

// ---------- fills ----------
__global__ void k_fill_f32(float* p, float v, long long n) {
    long long t = (long long)blockIdx.x * blockDim.x + threadIdx.x;
    if (t < n) p[t] = v;
}
__global__ void k_fill_u32(unsigned* p, unsigned v, long long n) {
    long long t = (long long)blockIdx.x * blockDim.x + threadIdx.x;
    if (t < n) p[t] = v;
}

// ---------- embedding gathers ----------
__global__ void k_gather_user(const int* uf, const float* emb, float* out, int bs) {
    long long t = (long long)blockIdx.x * blockDim.x + threadIdx.x;
    long long n = (long long)bs * 160;
    if (t >= n) return;
    int u = (int)(t / 160), k = (int)(t % 160);
    out[t] = emb[(long long)uf[u * 5 + k / 32] * 32 + (k % 32)];
}
__global__ void k_gather_item(const int* iid, const int* ir, const float* ts,
                              const float* emb, float* out, int ni) {
    long long t = (long long)blockIdx.x * blockDim.x + threadIdx.x;
    long long n = (long long)ni * 96;
    if (t >= n) return;
    int i = (int)(t / 96), k = (int)(t % 96);
    float v;
    if (k < 32)      v = emb[(long long)iid[i] * 32 + k];
    else if (k < 64) v = emb[(long long)ir[i] * 32 + (k - 32)];
    else             v = ts[(long long)i * 32 + (k - 64)];
    out[t] = v;
}
__global__ void k_gather_feat(const int* fid, const float* emb, float* out, int nf) {
    long long t = (long long)blockIdx.x * blockDim.x + threadIdx.x;
    long long n = (long long)nf * 32;
    if (t >= n) return;
    int i = (int)(t / 32), k = (int)(t % 32);
    out[t] = emb[(long long)fid[i] * 32 + k];
}

// ---------- WMMA GEMM: C[M,N] = A[M,K] * W[N,K]^T (+ bias[n]) ----------
// One wave32 per 16 x (16*TN) output strip; A fragments loaded once, reused TN times.
// Compile-time K (multiple of 32) -> fully unrolled chain of v_wmma_f32_16x16x32_f16.
// Requires M%16==0, N%(16*TN)==0, blockDim=32, EXEC all-ones (no divergence).
template <int K, int TN>
__global__ void k_wmma_gemm_nt(const float* __restrict__ A, const float* __restrict__ W,
                               const float* __restrict__ bias, float* __restrict__ C,
                               int N) {
    constexpr int NC = K / 32;                 // K-chunks
    int ntn  = N / (16 * TN);
    int tile = blockIdx.x;
    int tm = tile / ntn, tn0 = (tile % ntn) * TN;
    int lane = threadIdx.x & 31;
    int half = lane >> 4, r = lane & 15;

    // ---- load A fragments once (row m = r); pairs (e,e+1) are K-adjacent -> float2 ----
    const float* Arow = A + (long long)(tm * 16 + r) * K;
    v16h a[NC];
#pragma unroll
    for (int c0 = 0; c0 < NC; ++c0) {
#pragma unroll
        for (int e = 0; e < 16; e += 2) {
            // A 16x32 f16 layout: k = (e&7) + ((e>>3)<<4) + half*8   (ISA 7.12.2)
            int ka = c0 * 32 + (e & 7) + ((e >> 3) << 4) + half * 8;
            float2 f = *(const float2*)(Arow + ka);
            a[c0][e]     = (_Float16)f.x;
            a[c0][e + 1] = (_Float16)f.y;
        }
    }

#pragma unroll
    for (int tni = 0; tni < TN; ++tni) {
        int n16 = (tn0 + tni) * 16;
        const float* Wrow = W + (long long)(n16 + r) * K;   // B = W^T, col n = r
        v8f c = {};
#pragma unroll
        for (int c0 = 0; c0 < NC; ++c0) {
            v16h b;
#pragma unroll
            for (int e = 0; e < 16; e += 2) {
                // B 32x16 f16 layout: k = (e&7) + ((e>>3)<<4) + half*8
                int kb = c0 * 32 + (e & 7) + ((e >> 3) << 4) + half * 8;
                float2 f = *(const float2*)(Wrow + kb);
                b[e]     = (_Float16)f.x;
                b[e + 1] = (_Float16)f.y;
            }
            c = __builtin_amdgcn_wmma_f32_16x16x32_f16(false, a[c0], false, b, (short)0, c,
                                                       false, false);
        }
        int n = n16 + r;
        float bv = bias ? bias[n] : 0.f;     // hoisted: one branch per lane, not per store
#pragma unroll
        for (int v = 0; v < 8; ++v) {
            int m = tm * 16 + half * 8 + v;  // C: row = vgpr + 8*half, col = lane&15
            C[(long long)m * N + n] = c[v] + bv;
        }
    }
}

// ---------- wl/wr = attention-folded weights: wl[h,k] = sum_d al[h,d]*W[h*32+d,k] ----------
__global__ void k_make_wlr(const float* W, const float* al, const float* ar,
                           float* wl, float* wr, int H, int K) {
    int t = blockIdx.x * blockDim.x + threadIdx.x;
    if (t >= H * K) return;
    int h = t / K, k = t % K;
    float sl = 0.f, sr = 0.f;
    for (int d = 0; d < 32; ++d) {
        float w = W[(long long)(h * 32 + d) * K + k];
        sl += al[h * 32 + d] * w;
        sr += ar[h * 32 + d] * w;
    }
    wl[t] = sl; wr[t] = sr;
}

// ---------- el/er: out[n,h] = dot(X[n,:], wv[h,:]) ----------
__global__ void k_ler(const float* __restrict__ X, const float* __restrict__ wv,
                      float* out, int N, int H, int K) {
    long long t = (long long)blockIdx.x * blockDim.x + threadIdx.x;
    if (t >= (long long)N * H) return;
    int n = (int)(t / H), h = (int)(t % H);
    const float* x = X + (long long)n * K;
    const float* w = wv + (long long)h * K;
    float s = 0.f;
    for (int k = 0; k < K; ++k) s += x[k] * w[k];
    out[t] = s;
}

// ---------- edge phase ----------
__global__ void k_edge_e(const int* src, const int* dst, const float* el, const float* er,
                         float* e, unsigned* m, int E, int H) {
    long long t = (long long)blockIdx.x * blockDim.x + threadIdx.x;
    if (t >= (long long)E * H) return;
    int ed = (int)(t / H), h = (int)(t % H);
    float x = el[(long long)src[ed] * H + h] + er[(long long)dst[ed] * H + h];
    x = (x > 0.f) ? x : NEG_SLOPE * x;       // leaky relu
    e[t] = x;
    atomicMax(&m[(long long)dst[ed] * H + h], ford(x));
}
__global__ void k_edge_exp(const int* dst, float* e, const unsigned* m, float* den,
                           int E, int H) {
    long long t = (long long)blockIdx.x * blockDim.x + threadIdx.x;
    if (t >= (long long)E * H) return;
    int ed = (int)(t / H), h = (int)(t % H);
    float mm = funord(m[(long long)dst[ed] * H + h]);
    float ex = __expf(e[t] - mm);
    e[t] = ex;
    atomicAdd(&den[(long long)dst[ed] * H + h], ex);
}
__global__ void k_alpha(const int* dst, float* e, const float* den, int E, int H) {
    long long t = (long long)blockIdx.x * blockDim.x + threadIdx.x;
    if (t >= (long long)E * H) return;
    int ed = (int)(t / H), h = (int)(t % H);
    e[t] = e[t] / fmaxf(den[(long long)dst[ed] * H + h], 1e-9f);
}
__global__ void k_edge_scatter(const int* src, const int* dst, const float* __restrict__ alpha,
                               const float* __restrict__ fs, float* out, int E, int H, int D) {
    long long t = (long long)blockIdx.x * blockDim.x + threadIdx.x;
    int HD = H * D;
    if (t >= (long long)E * HD) return;
    int ed = (int)(t / HD), j = (int)(t % HD), h = j / D;
    atomicAdd(&out[(long long)dst[ed] * HD + j],
              fs[(long long)src[ed] * HD + j] * alpha[(long long)ed * H + h]);
}
__global__ void k_add_bias(float* out, const float* b, int N, int HD) {
    long long t = (long long)blockIdx.x * blockDim.x + threadIdx.x;
    if (t >= (long long)N * HD) return;
    out[t] += b[t % HD];
}

// ---------- final cosine + sigmoid ----------
__global__ void k_final(const float* hu, const float* hi, const int* tidx,
                        float* out, int bs) {
    int u = blockIdx.x * blockDim.x + threadIdx.x;
    if (u >= bs) return;
    const float* a = hu + (long long)u * 32;
    const float* b = hi + (long long)tidx[u] * 32;
    float na = 0.f, nb = 0.f, d = 0.f;
    for (int k = 0; k < 32; ++k) { na += a[k] * a[k]; nb += b[k] * b[k]; d += a[k] * b[k]; }
    float c = d / (fmaxf(sqrtf(na), 1e-8f) * fmaxf(sqrtf(nb), 1e-8f));
    out[u] = 1.f / (1.f + __expf(-c));
}

// =====================================================================
static inline unsigned cdiv(long long a, int b) { return (unsigned)((a + b - 1) / b); }

extern "C" void kernel_launch(void* const* d_in, const int* in_sizes, int n_in,
                              void* d_out, int out_size, void* d_ws, size_t ws_size,
                              hipStream_t stream) {
    // ---- inputs (setup_inputs dict order; params depth-first insertion order) ----
    const int*   user_feat = (const int*)  d_in[0];
    const int*   item_id   = (const int*)  d_in[1];
    const int*   item_rat  = (const int*)  d_in[2];
    const float* ts_emb    = (const float*)d_in[3];
    const int*   feat_id   = (const int*)  d_in[4];
    const int*   bt_src    = (const int*)  d_in[5];
    const int*   bt_dst    = (const int*)  d_in[6];
    const int*   hi_src    = (const int*)  d_in[7];
    const int*   hi_dst    = (const int*)  d_in[8];
    const int*   int_src   = (const int*)  d_in[9];
    const int*   int_dst   = (const int*)  d_in[10];
    const int*   cb_src    = (const int*)  d_in[11];
    const int*   cb_dst    = (const int*)  d_in[12];
    const int*   tgt_idx   = (const int*)  d_in[13];
    const float* emb_table = (const float*)d_in[14];
    const float* user_W    = (const float*)d_in[15];
    const float* user_b    = (const float*)d_in[16];
    const float* item_W    = (const float*)d_in[17];
    const float* item_b    = (const float*)d_in[18];
    const float* feat_W    = (const float*)d_in[19];
    const float* feat_b    = (const float*)d_in[20];
    // params1: belongto=21..24, hasinstance=25..28, interacted=29..32, clickby=33..36
    // params2: belongto=37..40 (unused), hasinstance=41..44, interacted=45..48, clickby=49..52
    const float* p1_bt_W  = (const float*)d_in[21]; const float* p1_bt_al = (const float*)d_in[22];
    const float* p1_bt_ar = (const float*)d_in[23]; const float* p1_bt_b  = (const float*)d_in[24];
    const float* p1_hi_W  = (const float*)d_in[25]; const float* p1_hi_al = (const float*)d_in[26];
    const float* p1_hi_ar = (const float*)d_in[27]; const float* p1_hi_b  = (const float*)d_in[28];
    const float* p1_in_W  = (const float*)d_in[29]; const float* p1_in_al = (const float*)d_in[30];
    const float* p1_in_ar = (const float*)d_in[31]; const float* p1_in_b  = (const float*)d_in[32];
    const float* p1_cb_W  = (const float*)d_in[33]; const float* p1_cb_al = (const float*)d_in[34];
    const float* p1_cb_ar = (const float*)d_in[35]; const float* p1_cb_b  = (const float*)d_in[36];
    const float* p2_hi_W  = (const float*)d_in[41]; const float* p2_hi_al = (const float*)d_in[42];
    const float* p2_hi_ar = (const float*)d_in[43]; const float* p2_hi_b  = (const float*)d_in[44];
    const float* p2_in_W  = (const float*)d_in[45]; const float* p2_in_al = (const float*)d_in[46];
    const float* p2_in_ar = (const float*)d_in[47]; const float* p2_in_b  = (const float*)d_in[48];
    const float* p2_cb_W  = (const float*)d_in[49]; const float* p2_cb_al = (const float*)d_in[50];
    const float* p2_cb_ar = (const float*)d_in[51]; const float* p2_cb_b  = (const float*)d_in[52];

    const int bs     = in_sizes[0] / 5;   // 4096
    const int n_item = in_sizes[1];       // 100000
    const int n_feat = in_sizes[4];       // 100000
    const int E_bt   = in_sizes[5];       // 300000
    const int E_int  = in_sizes[9];       // 204800

    // ---- workspace carve-up (floats) ----
    float* base = (float*)d_ws;
    long long o = 0;
    auto carve = [&](long long n) { float* p = base + o; o += n; return p; };
    float* ITEM_E = carve((long long)n_item * 32);
    float* FEAT_E = carve((long long)n_feat * 32);
    float* USER_E = carve((long long)bs * 32);
    float* H_FEAT = carve((long long)n_feat * 256);
    float* H_ITEM = carve((long long)n_item * 256);
    float* H_USER = carve((long long)bs * 256);
    float* FS     = carve((long long)n_item * 256);   // fs temp; also gather staging
    float* EL     = carve((long long)n_item * 8);
    float* ER     = carve((long long)n_item * 8);
    unsigned* MB  = (unsigned*)carve((long long)n_item * 8);
    float* DEN    = carve((long long)n_item * 8);
    float* EB     = carve((long long)E_bt * 8);
    float* WL     = carve(8 * 256);
    float* WR     = carve(8 * 256);
    float* H_IT2  = carve((long long)n_item * 32);
    float* H_US2  = carve((long long)bs * 32);
    (void)ws_size; (void)n_in; (void)out_size;

    const int TB = 256;

    // ---- generic GAT conv: accumulate into outbuf (pre-zeroed), then add bias ----
    auto gat = [&](const float* hsrc, int Ns, const float* hdst, int Nd,
                   const int* srcI, const int* dstI, int E,
                   const float* W, const float* al, const float* ar, const float* bias,
                   int H, int K, float* outbuf) {
        int HD = H * 32;
        k_make_wlr<<<cdiv(H * K, TB), TB, 0, stream>>>(W, al, ar, WL, WR, H, K);
        // fs = hsrc @ W^T   (WMMA, compile-time K)
        if (K == 32)   // layer 1: HD=256 -> 4 N-tiles per wave (A fragment reuse)
            k_wmma_gemm_nt<32, 4><<<(unsigned)((Ns / 16) * (HD / 64)), 32, 0, stream>>>(
                hsrc, W, nullptr, FS, HD);
        else           // layer 2: K=256, HD=32 -> 2 N-tiles per wave
            k_wmma_gemm_nt<256, 2><<<(unsigned)((Ns / 16) * (HD / 32)), 32, 0, stream>>>(
                hsrc, W, nullptr, FS, HD);
        k_ler<<<cdiv((long long)Ns * H, TB), TB, 0, stream>>>(hsrc, WL, EL, Ns, H, K);
        k_ler<<<cdiv((long long)Nd * H, TB), TB, 0, stream>>>(hdst, WR, ER, Nd, H, K);
        k_fill_u32<<<cdiv((long long)Nd * H, TB), TB, 0, stream>>>(MB, ORD_NEG_INF, (long long)Nd * H);
        k_fill_f32<<<cdiv((long long)Nd * H, TB), TB, 0, stream>>>(DEN, 0.f, (long long)Nd * H);
        k_edge_e  <<<cdiv((long long)E * H, TB), TB, 0, stream>>>(srcI, dstI, EL, ER, EB, MB, E, H);
        k_edge_exp<<<cdiv((long long)E * H, TB), TB, 0, stream>>>(dstI, EB, MB, DEN, E, H);
        k_alpha   <<<cdiv((long long)E * H, TB), TB, 0, stream>>>(dstI, EB, DEN, E, H);
        k_edge_scatter<<<cdiv((long long)E * HD, TB), TB, 0, stream>>>(srcI, dstI, EB, FS,
                                                                       outbuf, E, H, 32);
        k_add_bias<<<cdiv((long long)Nd * HD, TB), TB, 0, stream>>>(outbuf, bias, Nd, HD);
    };

    // ================= Phase 0: node embeddings (WMMA GEMMs, N=32 -> TN=2) =================
    k_gather_user<<<cdiv((long long)bs * 160, TB), TB, 0, stream>>>(user_feat, emb_table, FS, bs);
    k_wmma_gemm_nt<160, 2><<<(unsigned)(bs / 16), 32, 0, stream>>>(FS, user_W, user_b, USER_E, 32);

    k_gather_item<<<cdiv((long long)n_item * 96, TB), TB, 0, stream>>>(item_id, item_rat, ts_emb,
                                                                       emb_table, FS, n_item);
    k_wmma_gemm_nt<96, 2><<<(unsigned)(n_item / 16), 32, 0, stream>>>(FS, item_W, item_b, ITEM_E, 32);

    k_gather_feat<<<cdiv((long long)n_feat * 32, TB), TB, 0, stream>>>(feat_id, emb_table, FS, n_feat);
    k_wmma_gemm_nt<32, 2><<<(unsigned)(n_feat / 16), 32, 0, stream>>>(FS, feat_W, feat_b, FEAT_E, 32);

    // ================= Layer 1 (8 heads, K=32, HD=256) =================
    k_fill_f32<<<cdiv((long long)n_feat * 256, TB), TB, 0, stream>>>(H_FEAT, 0.f, (long long)n_feat * 256);
    k_fill_f32<<<cdiv((long long)n_item * 256, TB), TB, 0, stream>>>(H_ITEM, 0.f, (long long)n_item * 256);
    k_fill_f32<<<cdiv((long long)bs * 256, TB), TB, 0, stream>>>(H_USER, 0.f, (long long)bs * 256);

    gat(ITEM_E, n_item, FEAT_E, n_feat, bt_src, bt_dst, E_bt,
        p1_bt_W, p1_bt_al, p1_bt_ar, p1_bt_b, 8, 32, H_FEAT);
    gat(FEAT_E, n_feat, ITEM_E, n_item, hi_src, hi_dst, E_bt,
        p1_hi_W, p1_hi_al, p1_hi_ar, p1_hi_b, 8, 32, H_ITEM);
    gat(USER_E, bs, ITEM_E, n_item, int_src, int_dst, E_int,
        p1_in_W, p1_in_al, p1_in_ar, p1_in_b, 8, 32, H_ITEM);
    gat(ITEM_E, n_item, USER_E, bs, cb_src, cb_dst, E_int,
        p1_cb_W, p1_cb_al, p1_cb_ar, p1_cb_b, 8, 32, H_USER);

    // ================= Layer 2 (1 head, K=256, HD=32) =================
    k_fill_f32<<<cdiv((long long)n_item * 32, TB), TB, 0, stream>>>(H_IT2, 0.f, (long long)n_item * 32);
    k_fill_f32<<<cdiv((long long)bs * 32, TB), TB, 0, stream>>>(H_US2, 0.f, (long long)bs * 32);

    gat(H_FEAT, n_feat, H_ITEM, n_item, hi_src, hi_dst, E_bt,
        p2_hi_W, p2_hi_al, p2_hi_ar, p2_hi_b, 1, 256, H_IT2);
    gat(H_USER, bs, H_ITEM, n_item, int_src, int_dst, E_int,
        p2_in_W, p2_in_al, p2_in_ar, p2_in_b, 1, 256, H_IT2);
    gat(H_ITEM, n_item, H_USER, bs, cb_src, cb_dst, E_int,
        p2_cb_W, p2_cb_al, p2_cb_ar, p2_cb_b, 1, 256, H_US2);

    // ================= cosine similarity + sigmoid =================
    k_final<<<cdiv(bs, TB), TB, 0, stream>>>(H_US2, H_IT2, tgt_idx, (float*)d_out, bs);
}
